// SelfAttention_22308060136011
// MI455X (gfx1250) — matrix operand order repaired and checked by reading, not compiled, not run on hardware
//
#include <hip/hip_runtime.h>
#include <hip/hip_bf16.h>

#define BB 16
#define LL 2048
#define DD 512
#define AA 128

typedef __attribute__((ext_vector_type(16))) __bf16 v16bf;
typedef __attribute__((ext_vector_type(8)))  float  v8f;

union V16BF { v16bf v; unsigned int u[8]; };

__device__ __forceinline__ unsigned int packbf(float a, float b) {
  unsigned int xa = __float_as_uint(a);
  unsigned int xb = __float_as_uint(b);
  unsigned int ra = (xa + 0x7FFFu + ((xa >> 16) & 1u)) >> 16;
  unsigned int rb = (xb + 0x7FFFu + ((xb >> 16) & 1u)) >> 16;
  return ra | (rb << 16);
}

// ---------------------------------------------------------------------------
// K1: s[b,j] = query . tanh(x[b,j,:] @ w_a)   via WMMA bf16 (f32 accumulate)
// One wave = 16 rows (M=16). 8 N-tiles of 16 (A=128). K=512 in chunks of 32.
// ---------------------------------------------------------------------------
__global__ __launch_bounds__(256) void gemm_s_kernel(
    const float* __restrict__ x, const float* __restrict__ w_a,
    const float* __restrict__ query, float* __restrict__ s_out) {
  // w_a as bf16 in LDS, column-major [a][d], uint-packed pairs, stride 260 uints
  // (=520 bf16): lanes (distinct n) land 4 banks apart -> conflict-free.
  __shared__ unsigned int sB[AA * 260];
  const int tid = threadIdx.x;
  for (int idx = tid; idx < AA * (DD / 2); idx += 256) {
    int a  = idx >> 8;          // 0..127
    int dp = idx & 255;         // d pair index
    int d  = dp * 2;
    float f0 = w_a[(size_t)d * AA + a];
    float f1 = w_a[(size_t)(d + 1) * AA + a];
    sB[a * 260 + dp] = packbf(f0, f1);
  }
  __syncthreads();

  const int lane = tid & 31;
  const int wave = tid >> 5;
  const int ln   = lane & 15;   // M (for A) / N (for B,C)
  const int lh   = lane >> 4;   // half select
  const int tile = blockIdx.x * 8 + wave;
  const int row  = tile * 16 + ln;
  const float* xr = x + (size_t)row * DD;

  float qv[8];
#pragma unroll
  for (int nt = 0; nt < 8; ++nt) qv[nt] = query[nt * 16 + ln];

  v8f acc[8] = {};

  for (int kc = 0; kc < DD / 32; ++kc) {
    // A fragment: lane holds K = [kb, kb+8) and [kb+16, kb+24), kb = kc*32 + lh*8
    int kb = kc * 32 + lh * 8;
    float4 f0 = *(const float4*)(xr + kb);
    float4 f1 = *(const float4*)(xr + kb + 4);
    float4 f2 = *(const float4*)(xr + kb + 16);
    float4 f3 = *(const float4*)(xr + kb + 20);
    V16BF Am;
    Am.u[0] = packbf(f0.x, f0.y);
    Am.u[1] = packbf(f0.z, f0.w);
    Am.u[2] = packbf(f1.x, f1.y);
    Am.u[3] = packbf(f1.z, f1.w);
    Am.u[4] = packbf(f2.x, f2.y);
    Am.u[5] = packbf(f2.z, f2.w);
    Am.u[6] = packbf(f3.x, f3.y);
    Am.u[7] = packbf(f3.z, f3.w);

    const int kub = (kc * 32 + lh * 8) >> 1;  // uint offset in column
#pragma unroll
    for (int nt = 0; nt < 8; ++nt) {
      const int n = nt * 16 + ln;
      const unsigned int* bp = &sB[n * 260 + kub];
      V16BF Bm;
#pragma unroll
      for (int i = 0; i < 8; ++i) Bm.u[i] = bp[i + (i & 4)];
      acc[nt] = __builtin_amdgcn_wmma_f32_16x16x32_bf16(
          false, Am.v, false, Bm.v, (short)0, acc[nt], false, false);
    }
  }

  // C layout: (VGPR r, half h) -> row r + 8h, lane&15 -> N. Reduce over N.
#pragma unroll
  for (int r = 0; r < 8; ++r) {
    float p = 0.f;
#pragma unroll
    for (int nt = 0; nt < 8; ++nt) p += tanhf(acc[nt][r]) * qv[nt];
    p += __shfl_xor(p, 1, 16);
    p += __shfl_xor(p, 2, 16);
    p += __shfl_xor(p, 4, 16);
    p += __shfl_xor(p, 8, 16);
    if (ln == 0) s_out[tile * 16 + lh * 8 + r] = p;
  }
}

// ---------------------------------------------------------------------------
// K2: per batch, prefix-logsumexp over s -> P,Q; suffix sum of Q -> colsum.
//   P[j] = am_j * exp(s_j);  Q[i] = am_i * exp(-LSE_i)
//   colsum[j] = P[j] * sum_{i>=j} Q[i]
// 256 threads, 8 elements each; Hillis-Steele scan of thread totals.
// ---------------------------------------------------------------------------
__global__ __launch_bounds__(256) void scan_kernel(
    const float* __restrict__ s, const int* __restrict__ am,
    float* __restrict__ P, float* __restrict__ Q, float* __restrict__ colsum) {
  __shared__ float sv[LL];
  __shared__ float tm[256];
  __shared__ float tz[256];
  const int b   = blockIdx.x;
  const int tid = threadIdx.x;
  for (int i = tid; i < LL; i += 256) sv[i] = s[b * LL + i];
  __syncthreads();

  const int base = tid * 8;
  float lm[8], lz[8];
  float m = -INFINITY, z = 0.f;
#pragma unroll
  for (int e = 0; e < 8; ++e) {
    float v  = sv[base + e];
    float nm = fmaxf(m, v);
    z = z * __expf(m - nm) + __expf(v - nm);
    m = nm;
    lm[e] = m; lz[e] = z;
  }
  tm[tid] = m; tz[tid] = z;
  __syncthreads();
  for (int off = 1; off < 256; off <<= 1) {
    float pm = -INFINITY, pz = 0.f;
    if (tid >= off) { pm = tm[tid - off]; pz = tz[tid - off]; }
    __syncthreads();
    if (tid >= off) {
      float nm = fmaxf(m, pm);
      z = z * __expf(m - nm) + pz * __expf(pm - nm);
      m = nm;
      tm[tid] = m; tz[tid] = z;
    }
    __syncthreads();
  }
  float pem = -INFINITY, pez = 0.f;
  if (tid > 0) { pem = tm[tid - 1]; pez = tz[tid - 1]; }

  float lp[8], lq[8];
#pragma unroll
  for (int e = 0; e < 8; ++e) {
    float nm  = fmaxf(pem, lm[e]);
    float ez  = pez * __expf(pem - nm) + lz[e] * __expf(lm[e] - nm);
    float lse = nm + __logf(ez);
    int   j   = base + e;
    int   a   = am[b * LL + j];
    float pj  = a ? __expf(sv[j]) : 0.f;
    float qj  = a ? __expf(-lse) : 0.f;
    lp[e] = pj; lq[e] = qj;
    P[b * LL + j] = pj;
    Q[b * LL + j] = qj;
  }
  __syncthreads();  // before reusing tm for the add-scan

  float linc[8];
  float run = 0.f;
#pragma unroll
  for (int e = 0; e < 8; ++e) { run += lq[e]; linc[e] = run; }
  tm[tid] = run;
  __syncthreads();
  float t = run;
  for (int off = 1; off < 256; off <<= 1) {
    float pv = 0.f;
    if (tid >= off) pv = tm[tid - off];
    __syncthreads();
    if (tid >= off) { t += pv; tm[tid] = t; }
    __syncthreads();
  }
  float texcl = (tid > 0) ? tm[tid - 1] : 0.f;
  float total = tm[255];
#pragma unroll
  for (int e = 0; e < 8; ++e) {
    float incl = texcl + linc[e];
    float V    = total - incl + lq[e];  // suffix sum from this element
    colsum[b * LL + base + e] = lp[e] * V;
  }
}

// ---------------------------------------------------------------------------
// K3: dmat[b,i,j] = (j<=i) ? Q[b,i]*P[b,j] : 0. One block per (b,i) row.
// Pure bandwidth: 268 MB of float4 stores, P is L2-resident.
// ---------------------------------------------------------------------------
__global__ __launch_bounds__(256) void dmat_kernel(
    const float* __restrict__ P, const float* __restrict__ Q,
    float* __restrict__ dmat) {
  const int bi = blockIdx.x;
  const int b  = bi >> 11;
  const int i  = bi & (LL - 1);
  const float q = Q[bi];
  const float4* P4 = (const float4*)(P + (size_t)b * LL);
  float4* out = (float4*)(dmat + (size_t)bi * LL);
  const int tid = threadIdx.x;
#pragma unroll
  for (int k = 0; k < 2; ++k) {
    int j4 = tid + k * 256;
    int j  = j4 * 4;
    float4 p = P4[j4];
    float4 r;
    r.x = (j     <= i) ? q * p.x : 0.f;
    r.y = (j + 1 <= i) ? q * p.y : 0.f;
    r.z = (j + 2 <= i) ? q * p.z : 0.f;
    r.w = (j + 3 <= i) ? q * p.w : 0.f;
    out[j4] = r;
  }
}

// ---------------------------------------------------------------------------
// K4: a[b,j,d] = x[b,j,d] * colsum[b,j]
// ---------------------------------------------------------------------------
__global__ __launch_bounds__(256) void a_kernel(
    const float* __restrict__ x, const float* __restrict__ colsum,
    float* __restrict__ out) {
  size_t idx = (size_t)blockIdx.x * 256 + threadIdx.x;  // float4 index
  const float4* x4 = (const float4*)x;
  float4 v = x4[idx];
  size_t row = idx >> 7;  // 128 float4 per row (D=512)
  float c = colsum[row];
  float4 r = make_float4(v.x * c, v.y * c, v.z * c, v.w * c);
  ((float4*)out)[idx] = r;
}

extern "C" void kernel_launch(void* const* d_in, const int* in_sizes, int n_in,
                              void* d_out, int out_size, void* d_ws, size_t ws_size,
                              hipStream_t stream) {
  const float* x     = (const float*)d_in[0];  // (16,2048,512) f32
  const float* w_a   = (const float*)d_in[1];  // (512,128) f32
  const float* query = (const float*)d_in[2];  // (128,) f32
  const int*   am    = (const int*)d_in[3];    // (16,2048) i32

  float* ws      = (float*)d_ws;               // needs 4*16*2048 floats = 512 KB
  float* s       = ws;
  float* P       = ws + (size_t)BB * LL;
  float* Q       = P  + (size_t)BB * LL;
  float* colsum  = Q  + (size_t)BB * LL;

  float* out_a    = (float*)d_out;                         // (B,L,D)
  float* out_dmat = out_a + (size_t)BB * LL * DD;          // (B,L,L)

  gemm_s_kernel<<<(BB * LL) / (16 * 8), 256, 0, stream>>>(x, w_a, query, s);
  scan_kernel<<<BB, 256, 0, stream>>>(s, am, P, Q, colsum);
  dmat_kernel<<<BB * LL, 256, 0, stream>>>(P, Q, out_dmat);
  a_kernel<<<(BB * LL * DD / 4) / 256, 256, 0, stream>>>(x, colsum, out_a);
}
